// KCGE_83915071030129
// MI455X (gfx1250) — matrix operand order repaired and checked by reading, not compiled, not run on hardware
//
#include <hip/hip_runtime.h>
#include <hip/hip_bf16.h>

#define DIM  128
#define NREL 8
#define LDA  136    // padded LDS stride (bf16 elements) for the A tile

// meta layout (ints): [0..8) counts, [8..16) cursors, [16..24) padded edge starts,
//                     [24..33) tile offsets (tileOfs[NREL] = total tiles)
#define M_CNT 0
#define M_CUR 8
#define M_PST 16
#define M_TOF 24

typedef __attribute__((ext_vector_type(16))) __bf16       v16bf;
typedef __attribute__((ext_vector_type(8)))  float        v8f;
typedef __attribute__((ext_vector_type(8)))  unsigned int v8u;

static __device__ __forceinline__ unsigned short f2bf(float f) {
    unsigned u = __builtin_bit_cast(unsigned, f);
    unsigned r = u + 0x7FFFu + ((u >> 16) & 1u);   // round-to-nearest-even
    return (unsigned short)(r >> 16);
}

// ---------------------------------------------------------------------------
// Elementwise / normalization helpers
// ---------------------------------------------------------------------------
__global__ void zero_f32_kernel(float* __restrict__ p, long long n) {
    long long i = (long long)blockIdx.x * blockDim.x + threadIdx.x;
    if (i < n) p[i] = 0.0f;
}

__global__ void fill_i32_kernel(int* __restrict__ p, int n, int v) {
    int i = blockIdx.x * 256 + threadIdx.x;
    if (i < n) p[i] = v;
}

__global__ void deg_count_kernel(const int* __restrict__ colv, float* __restrict__ deg, int E) {
    int e = blockIdx.x * 256 + threadIdx.x;
    if (e < E) atomicAdd(&deg[colv[e]], 1.0f);
}

__global__ void rsqrt_kernel(float* __restrict__ deg, int n) {
    int i = blockIdx.x * 256 + threadIdx.x;
    if (i < n) { float d = deg[i]; deg[i] = (d > 0.0f) ? rsqrtf(d) : 0.0f; }
}

__global__ void coef_kernel(const int* __restrict__ rowv, const int* __restrict__ colv,
                            const float* __restrict__ w, const float* __restrict__ dis,
                            float* __restrict__ coef, int E) {
    int e = blockIdx.x * 256 + threadIdx.x;
    if (e < E) coef[e] = dis[rowv[e]] * dis[colv[e]] * w[e];
}

__global__ void cvt_bf16_kernel(const float* __restrict__ src, unsigned short* __restrict__ dst,
                                long long n) {
    long long i = (long long)blockIdx.x * blockDim.x + threadIdx.x;
    if (i < n) dst[i] = f2bf(src[i]);
}

__global__ void relu_bias_kernel(float* __restrict__ z, const float* __restrict__ b, long long n) {
    long long i = (long long)blockIdx.x * blockDim.x + threadIdx.x;
    if (i < n) z[i] = fmaxf(z[i] + b[i & (DIM - 1)], 0.0f);
}

__global__ void bias_kernel(float* __restrict__ z, const float* __restrict__ b, long long n) {
    long long i = (long long)blockIdx.x * blockDim.x + threadIdx.x;
    if (i < n) z[i] += b[i & (DIM - 1)];
}

__global__ void combine_kernel(const float* __restrict__ x, const float* __restrict__ z1,
                               const float* __restrict__ z2, float* __restrict__ out, long long n) {
    long long i = (long long)blockIdx.x * blockDim.x + threadIdx.x;
    if (i < n) {
        float a = x[i], b = z1[i], c = z2[i];
        out[i]     = (a + b + c) * 0.25f;          // z_star
        out[n + i] = (b + c) * (1.0f / 3.0f);      // z_sharp (LAMDA=1)
    }
}

// ---------------------------------------------------------------------------
// Edge bucketing by relation type (device-side, padded to 128-edge tiles)
// ---------------------------------------------------------------------------
__global__ void count_types_kernel(const int* __restrict__ et, int* __restrict__ meta, int E) {
    int e = blockIdx.x * 256 + threadIdx.x;
    if (e < E) atomicAdd(&meta[M_CNT + et[e]], 1);
}

__global__ void setup_kernel(int* __restrict__ meta) {
    if (blockIdx.x == 0 && threadIdx.x == 0) {
        int epad = 0, tiles = 0;
        for (int r = 0; r < NREL; ++r) {
            meta[M_PST + r] = epad;
            meta[M_TOF + r] = tiles;
            int t = (meta[M_CNT + r] + 127) >> 7;
            tiles += t;
            epad  += t << 7;
        }
        meta[M_TOF + NREL] = tiles;
    }
}

__global__ void scatter_perm_kernel(const int* __restrict__ et, int* __restrict__ meta,
                                    int* __restrict__ perm, int E) {
    int e = blockIdx.x * 256 + threadIdx.x;
    if (e < E) {
        int r = et[e];
        int pos = meta[M_PST + r] + atomicAdd(&meta[M_CUR + r], 1);
        perm[pos] = e;
    }
}

// ---------------------------------------------------------------------------
// Pack W[r][K=128][C=128] (f32) into per-lane WMMA B fragments (bf16).
// Fragment f = (r*4 + ks)*8 + nt covers K slab [ks*32, ks*32+32), cols [nt*16, nt*16+16).
// B layout: lane holds column nt*16 + (lane&15); element j holds row k = ks*32 + (lane>>4)*8 + (j&7) + (j>>3)*16.
// ---------------------------------------------------------------------------
__global__ void pack_w_kernel(const float* __restrict__ W, unsigned short* __restrict__ wpk) {
    int idx = blockIdx.x * 256 + threadIdx.x;   // 0 .. 131071  (8*4*8*32*16)
    int j    = idx & 15;
    int lane = (idx >> 4) & 31;
    int nt   = (idx >> 9) & 7;
    int ks   = (idx >> 12) & 3;
    int r    = (idx >> 14) & 7;
    int K = ks * 32 + (lane >> 4) * 8 + (j & 7) + (j >> 3) * 16;
    int C = nt * 16 + (lane & 15);
    wpk[idx] = f2bf(W[((size_t)r * DIM + K) * DIM + C]);
}

// ---------------------------------------------------------------------------
// Fused grouped-GEMM message passing:
//   for each 128-edge tile of relation r:
//     A = gather x[row[e]]      (128 x 128, bf16, staged in LDS)
//     M = A @ W[r]              (WMMA bf16 -> f32)
//     out[col[e], :] += coef[e] * M[e, :]   (atomic f32)
// No xW intermediate; x (bf16) and out (f32) stay L2-resident on MI455X.
// ---------------------------------------------------------------------------
__global__ __launch_bounds__(256) void fused_conv_kernel(
    const unsigned short* __restrict__ xb,    // [N][128] bf16 features
    const unsigned short* __restrict__ wpk,   // packed B fragments (all relations)
    const int*   __restrict__ perm,           // padded, type-sorted edge ids (-1 = pad)
    const int*   __restrict__ rowv,
    const int*   __restrict__ colv,
    const float* __restrict__ coef,
    const int*   __restrict__ meta,
    float*       __restrict__ out)            // [N][128] f32 accumulator (pre-zeroed)
{
    __shared__ unsigned short As[128 * LDA];
    __shared__ int   col_s[128];
    __shared__ float coef_s[128];

    const int b = blockIdx.x;
    if (b >= meta[M_TOF + NREL]) return;

    // Which relation does this tile belong to?
    int r = 0;
    #pragma unroll
    for (int i = 1; i < NREL; ++i) r += (b >= meta[M_TOF + i]) ? 1 : 0;
    const int ebase = meta[M_PST + r] + (b - meta[M_TOF + r]) * 128;

    const int t = threadIdx.x;

    // Stage A tile: 128 gathered feature rows (bf16) + per-edge col/coef.
    {
        int slot = t >> 1;
        int half = t & 1;
        int eidx = perm[ebase + slot];
        uint4* dst = (uint4*)(As + slot * LDA + half * 64);
        if (eidx >= 0) {
            const uint4* src = (const uint4*)(xb + (size_t)rowv[eidx] * DIM + half * 64);
            #pragma unroll
            for (int i = 0; i < 8; ++i) dst[i] = src[i];
        } else {
            uint4 z = make_uint4(0u, 0u, 0u, 0u);
            #pragma unroll
            for (int i = 0; i < 8; ++i) dst[i] = z;
        }
        if (half == 0) {
            col_s[slot]  = (eidx >= 0) ? colv[eidx] : -1;
            coef_s[slot] = (eidx >= 0) ? coef[eidx] : 0.0f;
        }
    }
    __syncthreads();

    const int wave  = t >> 5;
    const int lane  = t & 31;
    const int wm    = (wave & 3) * 32;     // wave edge-row offset
    const int wn    = (wave >> 2) * 64;    // wave out-dim offset
    const int lrow  = lane & 15;
    const int lhalf = lane >> 4;

    v8f acc[2][4] = {};

    #pragma unroll
    for (int ks = 0; ks < 4; ++ks) {
        // A fragments per ISA 16-bit layout: lanes 0-15 K{0..7,16..23}, lanes 16-31 +8.
        v16bf afr[2];
        #pragma unroll
        for (int tm = 0; tm < 2; ++tm) {
            const unsigned int* lp = (const unsigned int*)
                (As + (wm + tm * 16 + lrow) * LDA + ks * 32 + lhalf * 8);
            v8u a;
            a[0] = lp[0]; a[1] = lp[1]; a[2] = lp[2];  a[3] = lp[3];
            a[4] = lp[8]; a[5] = lp[9]; a[6] = lp[10]; a[7] = lp[11];
            afr[tm] = __builtin_bit_cast(v16bf, a);
        }
        #pragma unroll
        for (int tn = 0; tn < 4; ++tn) {
            int frag = (r * 4 + ks) * 8 + (wn >> 4) + tn;
            const v8u* bp = (const v8u*)(wpk + ((size_t)frag * 32 + lane) * 16);
            v16bf bfr = __builtin_bit_cast(v16bf, *bp);
            #pragma unroll
            for (int tm = 0; tm < 2; ++tm) {
                acc[tm][tn] = __builtin_amdgcn_wmma_f32_16x16x32_bf16(
                    false, afr[tm], false, bfr, (short)0, acc[tm][tn], false, false);
            }
        }
    }

    // Epilogue: scale by coef and atomically scatter to destination nodes.
    // C layout: VGPR i -> edge slot (lhalf*8 + i), column = lrow.
    #pragma unroll
    for (int tm = 0; tm < 2; ++tm) {
        #pragma unroll
        for (int tn = 0; tn < 4; ++tn) {
            int colg  = wn + tn * 16 + lrow;
            int sbase = wm + tm * 16 + lhalf * 8;
            #pragma unroll
            for (int i = 0; i < 8; ++i) {
                int slot = sbase + i;
                int dst  = col_s[slot];
                if (dst >= 0)
                    atomicAdd(&out[(size_t)dst * DIM + colg], acc[tm][tn][i] * coef_s[slot]);
            }
        }
    }
}

// ---------------------------------------------------------------------------
extern "C" void kernel_launch(void* const* d_in, const int* in_sizes, int n_in,
                              void* d_out, int out_size, void* d_ws, size_t ws_size,
                              hipStream_t stream)
{
    const float* x  = (const float*)d_in[0];
    const int*   ei = (const int*)d_in[1];
    const int*   et = (const int*)d_in[2];
    const float* ew = (const float*)d_in[3];
    const float* W1 = (const float*)d_in[4];
    const float* b1 = (const float*)d_in[5];
    const float* W2 = (const float*)d_in[6];
    const float* b2 = (const float*)d_in[7];

    const int N = in_sizes[0] / DIM;
    const int E = in_sizes[2];
    const int* rowv = ei;
    const int* colv = ei + E;

    char* base = (char*)d_ws;
    size_t off = 0;
    auto alloc = [&](size_t bytes) -> void* {
        void* p = base + off;
        off = (off + bytes + 255) & ~(size_t)255;
        return p;
    };
    const int permCap = E + NREL * 128;

    float*          deg  = (float*)alloc((size_t)N * 4);
    float*          coef = (float*)alloc((size_t)E * 4);
    unsigned short* xb   = (unsigned short*)alloc((size_t)N * DIM * 2);
    unsigned short* wpk1 = (unsigned short*)alloc(131072 * 2);
    unsigned short* wpk2 = (unsigned short*)alloc(131072 * 2);
    int*            meta = (int*)alloc(64 * 4);
    int*            perm = (int*)alloc((size_t)permCap * 4);
    float*          z1   = (float*)alloc((size_t)N * DIM * 4);
    float*          z2   = (float*)alloc((size_t)N * DIM * 4);

    const long long nd = (long long)N * DIM;
    dim3 b256(256);
    auto g1 = [](long long n) { return dim3((unsigned)((n + 255) / 256)); };

    // Accumulators + bucketing state
    zero_f32_kernel<<<g1(N),  b256, 0, stream>>>(deg, N);
    zero_f32_kernel<<<g1(nd), b256, 0, stream>>>(z1, nd);
    zero_f32_kernel<<<g1(nd), b256, 0, stream>>>(z2, nd);
    fill_i32_kernel<<<1, 256, 0, stream>>>(meta, 64, 0);
    fill_i32_kernel<<<g1(permCap), b256, 0, stream>>>(perm, permCap, -1);

    // Degree normalization coefficients
    deg_count_kernel<<<g1(E), b256, 0, stream>>>(colv, deg, E);
    rsqrt_kernel<<<g1(N), b256, 0, stream>>>(deg, N);
    coef_kernel<<<g1(E), b256, 0, stream>>>(rowv, colv, ew, deg, coef, E);

    // Bucket edges by relation (padded to 128-edge tiles)
    count_types_kernel<<<g1(E), b256, 0, stream>>>(et, meta, E);
    setup_kernel<<<1, 1, 0, stream>>>(meta);
    scatter_perm_kernel<<<g1(E), b256, 0, stream>>>(et, meta, perm, E);

    // bf16 features + packed weight fragments
    cvt_bf16_kernel<<<g1(nd), b256, 0, stream>>>(x, xb, nd);
    pack_w_kernel<<<dim3(512), b256, 0, stream>>>(W1, wpk1);
    pack_w_kernel<<<dim3(512), b256, 0, stream>>>(W2, wpk2);

    const unsigned maxTiles = (unsigned)((E + 127) / 128 + NREL);

    // Layer 1: z1 = relu(scatter(coef * (x[row] @ W1[type])) + b1)
    fused_conv_kernel<<<dim3(maxTiles), b256, 0, stream>>>(
        xb, wpk1, perm, rowv, colv, coef, meta, z1);
    relu_bias_kernel<<<g1(nd), b256, 0, stream>>>(z1, b1, nd);

    // Layer 2: z2 = scatter(coef * (z1[row] @ W2[type])) + b2
    cvt_bf16_kernel<<<g1(nd), b256, 0, stream>>>(z1, xb, nd);
    fused_conv_kernel<<<dim3(maxTiles), b256, 0, stream>>>(
        xb, wpk2, perm, rowv, colv, coef, meta, z2);
    bias_kernel<<<g1(nd), b256, 0, stream>>>(z2, b2, nd);

    // Outputs: z_star then z_sharp
    combine_kernel<<<g1(nd), b256, 0, stream>>>(x, z1, z2, (float*)d_out, nd);

    (void)n_in; (void)out_size; (void)ws_size;
}
